// CorrelationModule_73383811220019
// MI455X (gfx1250) — compile-verified, matrix operand order checked by reading, not run on hardware
//
#include <hip/hip_runtime.h>
#include <hip/hip_bf16.h>

// ---------------------------------------------------------------------------
// CDNA5 (gfx1250) implementation of the DICL correlation module.
// GEMM-shaped convolutions run through v_wmma_f32_16x16x32_bf16 with fp32
// accumulation. Workgroups cooperatively stage im2col A-fragments in a
// double-buffered LDS slab (shared across all output-channel tiles); each
// wave register-blocks 4 N-tiles (4 WMMAs per K-step). Batchnorm statistics
// are fused into the WMMA epilogue.
// ---------------------------------------------------------------------------

typedef __bf16 bf16_t;
typedef bf16_t v16bf __attribute__((ext_vector_type(16)));
typedef float  v8f   __attribute__((ext_vector_type(8)));

union BF16Frag { v16bf v; unsigned short u[16]; };
union F32Acc   { v8f   v; float f[8]; };

__device__ __forceinline__ unsigned short f32_to_bf16(float f) {
  unsigned int u = __float_as_uint(f);
  u += 0x7FFFu + ((u >> 16) & 1u);          // round-to-nearest-even
  return (unsigned short)(u >> 16);
}
__device__ __forceinline__ float bf16_to_f32(unsigned short h) {
  return __uint_as_float(((unsigned int)h) << 16);
}

// K index held by (lane-half, element j) in the 16-bit 16x32 A fragment
// (ISA 7.12.2): lane<16 -> K {0..7, 16..23}; lane>=16 -> K {8..15, 24..31}.
// B fragments use the mirrored mapping with N = lane&15.
__device__ __forceinline__ int kmap(int khalf, int j) {
  return ((j < 8) ? j : j + 8) + 8 * khalf;
}

// ---------------------------------------------------------------------------
// Weight packing: fp32 OIHW -> bf16 B-fragment order.
// Layout: bp[((kt*Ntiles + nt)*32 + lane)*16 + j], zero padded past Cout/Ktot.
// ---------------------------------------------------------------------------
__global__ void pack_wB_kernel(const float* __restrict__ w,
                               unsigned short* __restrict__ bp,
                               int Cout, int Ktot, int KtSteps, int Ntiles) {
  int idx = blockIdx.x * blockDim.x + threadIdx.x;
  int total = KtSteps * Ntiles * 512;
  if (idx >= total) return;
  int j    = idx & 15;
  int lane = (idx >> 4) & 31;
  int t    = idx >> 9;
  int nt   = t % Ntiles;
  int kt   = t / Ntiles;
  int n = nt * 16 + (lane & 15);
  int k = kt * 32 + kmap(lane >> 4, j);
  float v = (n < Cout && k < Ktot) ? w[(size_t)n * Ktot + k] : 0.f;
  bp[idx] = f32_to_bf16(v);
}

// ---------------------------------------------------------------------------
// Stack build: bilinear-sample f2 at 81 displacements, concat [f1, f2s, delta]
// -> bf16 (162, 66, 64, 64). One thread per (n, pixel).
// ---------------------------------------------------------------------------
__global__ void build_stack_kernel(const float* __restrict__ f1,
                                   const float* __restrict__ f2,
                                   const float* __restrict__ coords,
                                   unsigned short* __restrict__ stack) {
  const int P = 4096, W = 64, H = 64, C = 32;
  int idx = blockIdx.x * blockDim.x + threadIdx.x;   // n*P + p
  if (idx >= 162 * P) return;
  int p = idx & (P - 1);
  int n = idx >> 12;
  int b = n / 81, d = n - b * 81;
  int di = d / 9, dj = d - di * 9;

  float sx = coords[((size_t)(b * 2 + 0)) * P + p] + (float)(di - 4);
  float sy = coords[((size_t)(b * 2 + 1)) * P + p] + (float)(dj - 4);
  float x0f = floorf(sx), y0f = floorf(sy);
  int x0 = (int)x0f, y0 = (int)y0f;
  float wx1 = sx - x0f, wx0 = 1.f - wx1;
  float wy1 = sy - y0f, wy0 = 1.f - wy1;
  bool vx0 = (x0 >= 0) & (x0 < W);
  bool vx1 = (x0 + 1 >= 0) & (x0 + 1 < W);
  bool vy0 = (y0 >= 0) & (y0 < H);
  bool vy1 = (y0 + 1 >= 0) & (y0 + 1 < H);
  int cx0 = min(max(x0, 0), W - 1), cx1 = min(max(x0 + 1, 0), W - 1);
  int cy0 = min(max(y0, 0), H - 1), cy1 = min(max(y0 + 1, 0), H - 1);

  const float* f1b = f1 + (size_t)b * C * P;
  const float* f2b = f2 + (size_t)b * C * P;
  unsigned short* st = stack + (size_t)n * 66 * P + p;

  for (int c = 0; c < C; ++c) {
    st[(size_t)c * P] = f32_to_bf16(f1b[(size_t)c * P + p]);
    const float* img = f2b + (size_t)c * P;
    float v00 = (vy0 && vx0) ? img[cy0 * W + cx0] : 0.f;
    float v01 = (vy0 && vx1) ? img[cy0 * W + cx1] : 0.f;
    float v10 = (vy1 && vx0) ? img[cy1 * W + cx0] : 0.f;
    float v11 = (vy1 && vx1) ? img[cy1 * W + cx1] : 0.f;
    float val = wy0 * (wx0 * v00 + wx1 * v01) + wy1 * (wx0 * v10 + wx1 * v11);
    st[(size_t)(32 + c) * P] = f32_to_bf16(val);
  }
  st[(size_t)64 * P] = f32_to_bf16((float)(di - 4));
  st[(size_t)65 * P] = f32_to_bf16((float)(dj - 4));
}

// ---------------------------------------------------------------------------
// Workgroup-cooperative implicit-GEMM conv via WMMA bf16->f32.
//
// Workgroup = 2*ceil(NT/4) waves covering a 32-pixel x Cout output block:
//   wave (mi, wcol) computes M-tile mi (16 pixels) x N-tiles {4*wcol..4*wcol+3}
//   (4 accumulators; the LDS A-fragment is reused 4x per K-step).
// Per 32-wide K-step:
//   - all threads cooperatively gather the next 32x32 im2col slab into the
//     other half of a double-buffered LDS slab (one barrier per K-step),
//   - each wave reads A with one contiguous 32B LDS load, loads 4 contiguous
//     B fragments (prefetching the next K-step), issues 4 WMMAs.
// k -> (input offset, ci, ky, kx) is precomputed in an LDS table: no integer
// division in the hot loop. Epilogue: optional bias/ReLU, bf16 store, fused
// per-channel sum / sum-of-squares atomics for batchnorm.
// ---------------------------------------------------------------------------
#define MAX_KTOT 1152

__global__ void __launch_bounds__(256) conv_wmma_kernel(
    const unsigned short* __restrict__ in, const unsigned short* __restrict__ wB,
    unsigned short* __restrict__ out, const float* __restrict__ bias,
    float* __restrict__ ch_sum, float* __restrict__ ch_ssq,
    int Nb, int Cin, int Hin, int Win, int Cout, int Hout, int Wout,
    int KH, int KW, int stride, int pad, int lhsdil, int KtSteps, int Ntiles,
    int relu_flag) {
  __shared__ int s_off[MAX_KTOT];                  // ci*HWi + ky*Win + kx
  __shared__ int s_meta[MAX_KTOT];                 // (ci<<8)|(ky<<4)|kx
  __shared__ unsigned short s_frag[2][2][32][16];  // [buf][mtile][lane][j]

  const int tid  = threadIdx.x;
  const int bd   = blockDim.x;
  const int lane = tid & 31;
  const int wid  = tid >> 5;
  const int nwcol = bd >> 6;                       // wave columns
  const int mi   = (wid >= nwcol) ? 1 : 0;
  const int wcol = wid - mi * nwcol;
  const int ni0  = wcol * 4;

  const int HWo  = Hout * Wout;
  const int HWi  = Hin * Win;
  const int Mtot = Nb * HWo;
  const int Ktot = Cin * KH * KW;
  const int KHW  = KH * KW;

  // k-decode table (runtime divisions only here, once per k)
  for (int k = tid; k < Ktot; k += bd) {
    int ci  = k / KHW;
    int rem = k - ci * KHW;
    int ky  = rem / KW, kx = rem - ky * KW;
    s_off[k]  = ci * HWi + ky * Win + kx;
    s_meta[k] = (ci << 8) | (ky << 4) | kx;
  }

  // gather role: this thread feeds output pixel mloc of the 32-pixel block,
  // handling k = kstart, kstart+kstride, ... within each 32-wide K-step.
  const int mloc    = tid & 31;
  const int kstart  = tid >> 5;
  const int kstride = bd >> 5;
  const int gmi  = mloc >> 4;
  const int grow = mloc & 15;
  int gm = blockIdx.x * 32 + mloc;
  bool gm_ok = gm < Mtot;
  int gmc = gm_ok ? gm : 0;
  int gn  = gmc / HWo;
  int gp  = gmc - gn * HWo;
  int gho = gp / Wout, gwo = gp - gho * Wout;
  const unsigned short* inb = in + (size_t)gn * Cin * HWi;
  const int gbase = (gho * stride - pad) * Win + (gwo * stride - pad);

  // cooperative im2col of K-step `kt` into LDS slab `buf`
  auto gather = [&](int kt, int buf) {
    for (int kk = kstart; kk < 32; kk += kstride) {
      int k = kt * 32 + kk;
      unsigned short v = 0;
      if (gm_ok && k < Ktot) {
        int meta = s_meta[k];
        int ky = (meta >> 4) & 15, kx = meta & 15;
        if (!lhsdil) {
          int hi = gho * stride + ky - pad;
          int wi = gwo * stride + kx - pad;
          if (hi >= 0 && hi < Hin && wi >= 0 && wi < Win)
            v = inb[s_off[k] + gbase];
        } else {                         // ConvTranspose(k=4,s=2,p=1)
          int yy = gho + ky - pad, xx = gwo + kx - pad;
          if (((yy | xx) >= 0) && !((yy | xx) & 1)) {
            int hi = yy >> 1, wi = xx >> 1;
            if (hi < Hin && wi < Win)
              v = inb[(meta >> 8) * HWi + hi * Win + wi];
          }
        }
      }
      // inverse fragment map: lane = row + 16*khalf, j = (k&7) + 8*(k>>4)
      int flane = grow + 16 * ((kk >> 3) & 1);
      int fj    = (kk & 7) + 8 * (kk >> 4);
      s_frag[buf][gmi][flane][fj] = v;
    }
  };

  F32Acc acc[4];
#pragma unroll
  for (int q = 0; q < 4; ++q)
#pragma unroll
    for (int i = 0; i < 8; ++i) acc[q].f[i] = 0.f;

  __syncthreads();            // table ready
  gather(0, 0);               // prologue slab
  __syncthreads();

  for (int kt = 0; kt < KtSteps; ++kt) {
    const int cb = kt & 1;

    // ---- WMMA: A from LDS (one 32B load, reused 4x), 4 B tiles per wave ----
    BF16Frag a;
    a.v = ((const BF16Frag*)&s_frag[cb][mi][lane][0])->v;
    const BF16Frag* bp =
        (const BF16Frag*)wB + ((size_t)kt * Ntiles + ni0) * 32 + lane;
    __builtin_prefetch((const void*)(bp + (size_t)Ntiles * 32), 0, 3);
#pragma unroll
    for (int q = 0; q < 4; ++q) {
      if (ni0 + q < Ntiles) {                 // wave-uniform guard
        BF16Frag b; b.v = bp[(size_t)q * 32].v;
        acc[q].v = __builtin_amdgcn_wmma_f32_16x16x32_bf16(
            false, a.v, false, b.v, (short)0, acc[q].v, false, false);
      }
    }

    // ---- gather next slab into the other buffer, single barrier ----
    if (kt + 1 < KtSteps) gather(kt + 1, cb ^ 1);
    __syncthreads();
  }

  // ---- epilogue: bias / ReLU / bf16 store / fused BN statistics ----
  const int khalf = lane >> 4;
  const int mrow  = lane & 15;
#pragma unroll
  for (int q = 0; q < 4; ++q) {
    int ni = ni0 + q;
    if (ni >= Ntiles) continue;
    int co = ni * 16 + mrow;
    bool co_ok = co < Cout;
    float bs = (bias && co_ok) ? bias[co] : 0.f;
    float ssum = 0.f, ssq = 0.f;
#pragma unroll
    for (int r = 0; r < 8; ++r) {
      int m = blockIdx.x * 32 + mi * 16 + r + 8 * khalf;
      if (!co_ok || m >= Mtot) continue;
      int n = m / HWo, p = m - n * HWo;
      float v = acc[q].f[r] + bs;
      ssum += v; ssq += v * v;
      if (relu_flag && v < 0.f) v = 0.f;
      out[((size_t)n * Cout + co) * HWo + p] = f32_to_bf16(v);
    }
    if (ch_sum && co_ok) {
      atomicAdd(&ch_sum[co], ssum);
      atomicAdd(&ch_ssq[co], ssq);
    }
  }
}

// ---------------------------------------------------------------------------
// Batchnorm (training-mode stats) + ReLU, applied in place on bf16.
// ---------------------------------------------------------------------------
__global__ void bn_relu_kernel(unsigned short* __restrict__ buf,
                               const float* __restrict__ sum,
                               const float* __restrict__ ssq,
                               const float* __restrict__ gamma,
                               const float* __restrict__ beta,
                               int Cout, int HW, int total, float inv_count) {
  int idx = blockIdx.x * blockDim.x + threadIdx.x;
  if (idx >= total) return;
  int c = (idx / HW) % Cout;
  float m   = sum[c] * inv_count;
  float var = ssq[c] * inv_count - m * m;
  float y = bf16_to_f32(buf[idx]);
  float o = gamma[c] * (y - m) * rsqrtf(var + 1e-5f) + beta[c];
  buf[idx] = f32_to_bf16(o > 0.f ? o : 0.f);
}

// ---------------------------------------------------------------------------
// conv5: 32 -> 1 channel, 3x3 (0.15% of FLOPs: scalar). fp32 cost (162,4096).
// ---------------------------------------------------------------------------
__global__ void conv5_kernel(const unsigned short* __restrict__ xd,
                             const float* __restrict__ w5,
                             const float* __restrict__ b5,
                             float* __restrict__ cost) {
  int idx = blockIdx.x * blockDim.x + threadIdx.x;   // n*4096 + p
  if (idx >= 162 * 4096) return;
  int p = idx & 4095, n = idx >> 12;
  int h = p >> 6, w = p & 63;
  float acc = b5[0];
  const unsigned short* xb = xd + (size_t)n * 32 * 4096;
  for (int ci = 0; ci < 32; ++ci) {
    const unsigned short* xc = xb + (size_t)ci * 4096;
    const float* wc = w5 + ci * 9;
#pragma unroll
    for (int ky = 0; ky < 3; ++ky) {
      int hi = h + ky - 1;
      if (hi < 0 || hi >= 64) continue;
#pragma unroll
      for (int kx = 0; kx < 3; ++kx) {
        int wi = w + kx - 1;
        if (wi < 0 || wi >= 64) continue;
        acc += bf16_to_f32(xc[hi * 64 + wi]) * wc[ky * 3 + kx];
      }
    }
  }
  cost[idx] = acc;
}

// ---------------------------------------------------------------------------
// DAP (81x81 matvec) + softmax over displacements + embedding reduction.
// dapw staged in LDS; one thread per (b, pixel). Writes final output
// (B, 81 + 32, H, W) fp32: cost passthrough + softmax-weighted embedding.
// ---------------------------------------------------------------------------
__global__ void dap_emb_kernel(const float* __restrict__ cost,
                               const unsigned short* __restrict__ e3,
                               const float* __restrict__ dapw,
                               float* __restrict__ outp) {
  __shared__ float sdap[81 * 81];
  for (int i = threadIdx.x; i < 81 * 81; i += blockDim.x) sdap[i] = dapw[i];
  __syncthreads();

  int idx = blockIdx.x * blockDim.x + threadIdx.x;   // b*4096 + p
  if (idx >= 2 * 4096) return;
  int p = idx & 4095, b = idx >> 12;

  float cd[81];
  for (int d = 0; d < 81; ++d)
    cd[d] = cost[((size_t)(b * 81 + d) << 12) + p];

  float s[81];
  float mx = -1e30f;
  for (int o = 0; o < 81; ++o) {
    const float* row = sdap + o * 81;
    float a = 0.f;
    for (int d = 0; d < 81; ++d) a += row[d] * cd[d];
    s[o] = a;
    mx = fmaxf(mx, a);
  }
  float se = 0.f;
  for (int o = 0; o < 81; ++o) { s[o] = __expf(s[o] - mx); se += s[o]; }
  float inv = 1.f / se;

  float emb[32];
#pragma unroll
  for (int c = 0; c < 32; ++c) emb[c] = 0.f;
  for (int o = 0; o < 81; ++o) {
    float wgt = s[o] * inv;
    const unsigned short* ev = e3 + (((size_t)(b * 81 + o) * 32) << 12) + p;
    for (int c = 0; c < 32; ++c) emb[c] += wgt * bf16_to_f32(ev[(size_t)c << 12]);
  }

  float* ob = outp + (size_t)b * 113 * 4096 + p;
  for (int d = 0; d < 81; ++d) ob[(size_t)d << 12] = cd[d];
  for (int c = 0; c < 32; ++c) ob[(size_t)(81 + c) << 12] = emb[c];
}

// ---------------------------------------------------------------------------
// Host orchestration (graph-capture safe: kernels + hipMemsetAsync only).
// ---------------------------------------------------------------------------
extern "C" void kernel_launch(void* const* d_in, const int* in_sizes, int n_in,
                              void* d_out, int out_size, void* d_ws, size_t ws_size,
                              hipStream_t stream) {
  (void)in_sizes; (void)n_in; (void)out_size; (void)ws_size;
  const float* f1  = (const float*)d_in[0];
  const float* f2  = (const float*)d_in[1];
  const float* co  = (const float*)d_in[2];
  const float* w1  = (const float*)d_in[3];  const float* g1 = (const float*)d_in[4];  const float* b1 = (const float*)d_in[5];
  const float* w2  = (const float*)d_in[6];  const float* g2 = (const float*)d_in[7];  const float* b2 = (const float*)d_in[8];
  const float* w3  = (const float*)d_in[9];  const float* g3 = (const float*)d_in[10]; const float* b3 = (const float*)d_in[11];
  const float* w4  = (const float*)d_in[12]; const float* g4 = (const float*)d_in[13]; const float* b4 = (const float*)d_in[14];
  const float* wd  = (const float*)d_in[15]; const float* gd = (const float*)d_in[16]; const float* bd = (const float*)d_in[17];
  const float* w5  = (const float*)d_in[18]; const float* b5 = (const float*)d_in[19];
  const float* e1w = (const float*)d_in[20]; const float* e1b = (const float*)d_in[21];
  const float* e2w = (const float*)d_in[22]; const float* e2b = (const float*)d_in[23];
  const float* e3w = (const float*)d_in[24]; const float* e3b = (const float*)d_in[25];
  const float* dapw = (const float*)d_in[26];

  char* ws = (char*)d_ws;

  // Workspace arena with lifetime-based aliasing (peak ~267 MiB):
  const size_t STACK_B = (size_t)162 * 66 * 4096 * 2;
  const size_t EMB_B   = (size_t)162 * 32 * 4096 * 2;
  const size_t E1_B    = (size_t)162 * 48 * 4096 * 2;
  const size_t E2_B    = (size_t)162 * 64 * 4096 * 2;

  const size_t OFF_STACK = 0;                            // dead after conv1
  const size_t OFF_EMB   = STACK_B;                      // persistent (e3 out)
  const size_t OFF_E1    = OFF_EMB + EMB_B;              // dead after e2
  const size_t OFF_E2    = OFF_E1 + E1_B;                // dead after e3
  const size_t OFF_X1    = OFF_E1;                       // aliases e1+e2
  const size_t OFF_X2    = 0;                            // aliases stack
  const size_t OFF_X3    = (size_t)162 * 128 * 1024 * 2; // after x2
  const size_t OFF_X4    = 0;                            // aliases x2
  const size_t OFF_XD    = (size_t)162 * 64 * 1024 * 2;  // after x4
  const size_t OFF_COST  = OFF_XD + (size_t)162 * 32 * 4096 * 2; // fp32

  size_t cur = OFF_E2 + E2_B;                            // persistent region
  auto carve = [&](size_t bytes) { size_t o = cur; cur += (bytes + 255) & ~(size_t)255; return o; };
  const size_t OFF_PW1 = carve((size_t)19 * 6 * 512 * 2);
  const size_t OFF_PW2 = carve((size_t)27 * 8 * 512 * 2);
  const size_t OFF_PW3 = carve((size_t)36 * 8 * 512 * 2);
  const size_t OFF_PW4 = carve((size_t)36 * 4 * 512 * 2);
  const size_t OFF_PWD = carve((size_t)32 * 2 * 512 * 2);
  const size_t OFF_PE1 = carve((size_t)3 * 3 * 512 * 2);
  const size_t OFF_PE2 = carve((size_t)2 * 4 * 512 * 2);
  const size_t OFF_PE3 = carve((size_t)2 * 2 * 512 * 2);
  const size_t OFF_SUM = carve(512);
  const size_t OFF_SSQ = carve(512);

  auto pack = [&](const float* w, size_t off, int Cout, int Ktot) {
    int Kt = (Ktot + 31) / 32, NT = (Cout + 15) / 16;
    int total = Kt * NT * 512;
    pack_wB_kernel<<<(total + 255) / 256, 256, 0, stream>>>(
        w, (unsigned short*)(ws + off), Cout, Ktot, Kt, NT);
  };
  auto conv = [&](size_t inOff, size_t wOff, size_t outOff, const float* bias,
                  bool stats, int relu, int Nb, int Cin, int Hin, int Win,
                  int Cout, int Hout, int Wout, int KH, int KW, int stride,
                  int pad, int lhsdil) {
    int Mtot = Nb * Hout * Wout;
    int MT = (Mtot + 31) / 32;                 // 32 pixels per workgroup
    int NT = (Cout + 15) / 16;
    int Kt = (Cin * KH * KW + 31) / 32;
    int bd = 64 * ((NT + 3) / 4);              // 2 M-waves per wave-column
    conv_wmma_kernel<<<dim3(MT), bd, 0, stream>>>(
        (const unsigned short*)(ws + inOff), (const unsigned short*)(ws + wOff),
        (unsigned short*)(ws + outOff), bias,
        stats ? (float*)(ws + OFF_SUM) : nullptr,
        stats ? (float*)(ws + OFF_SSQ) : nullptr,
        Nb, Cin, Hin, Win, Cout, Hout, Wout, KH, KW, stride, pad, lhsdil,
        Kt, NT, relu);
  };
  auto bnrelu = [&](size_t bufOff, const float* gamma, const float* beta,
                    int Cout, int HW) {
    int total = 162 * Cout * HW;
    bn_relu_kernel<<<(total + 255) / 256, 256, 0, stream>>>(
        (unsigned short*)(ws + bufOff), (const float*)(ws + OFF_SUM),
        (const float*)(ws + OFF_SSQ), gamma, beta, Cout, HW, total,
        1.f / (162.f * (float)HW));
  };

  // --- weight packing (bf16 B-fragments) ---
  pack(w1, OFF_PW1, 96, 66 * 9);
  pack(w2, OFF_PW2, 128, 96 * 9);
  pack(w3, OFF_PW3, 128, 128 * 9);
  pack(w4, OFF_PW4, 64, 128 * 9);
  pack(wd, OFF_PWD, 32, 64 * 16);
  pack(e1w, OFF_PE1, 48, 66);
  pack(e2w, OFF_PE2, 64, 48);
  pack(e3w, OFF_PE3, 32, 64);

  // --- stack build (grid_sample + concat) ---
  build_stack_kernel<<<(162 * 4096 + 255) / 256, 256, 0, stream>>>(
      f1, f2, co, (unsigned short*)(ws + OFF_STACK));

  // --- PairEmbedding 1x1 MLP (WMMA GEMMs) ---
  conv(OFF_STACK, OFF_PE1, OFF_E1, e1b, false, 1, 162, 66, 64, 64, 48, 64, 64, 1, 1, 1, 0, 0);
  conv(OFF_E1,    OFF_PE2, OFF_E2, e2b, false, 1, 162, 48, 64, 64, 64, 64, 64, 1, 1, 1, 0, 0);
  conv(OFF_E2,    OFF_PE3, OFF_EMB, e3b, false, 0, 162, 64, 64, 64, 32, 64, 64, 1, 1, 1, 0, 0);

  // --- MatchingNet (WMMA conv + fused BN stats + BN/ReLU pass) ---
  hipMemsetAsync(ws + OFF_SUM, 0, 1024, stream);
  conv(OFF_STACK, OFF_PW1, OFF_X1, nullptr, true, 0, 162, 66, 64, 64, 96, 64, 64, 3, 3, 1, 1, 0);
  bnrelu(OFF_X1, g1, b1, 96, 4096);

  hipMemsetAsync(ws + OFF_SUM, 0, 1024, stream);
  conv(OFF_X1, OFF_PW2, OFF_X2, nullptr, true, 0, 162, 96, 64, 64, 128, 32, 32, 3, 3, 2, 1, 0);
  bnrelu(OFF_X2, g2, b2, 128, 1024);

  hipMemsetAsync(ws + OFF_SUM, 0, 1024, stream);
  conv(OFF_X2, OFF_PW3, OFF_X3, nullptr, true, 0, 162, 128, 32, 32, 128, 32, 32, 3, 3, 1, 1, 0);
  bnrelu(OFF_X3, g3, b3, 128, 1024);

  hipMemsetAsync(ws + OFF_SUM, 0, 1024, stream);
  conv(OFF_X3, OFF_PW4, OFF_X4, nullptr, true, 0, 162, 128, 32, 32, 64, 32, 32, 3, 3, 1, 1, 0);
  bnrelu(OFF_X4, g4, b4, 64, 1024);

  // ConvTranspose2d(k=4,s=2,p=1) == lhs-dilated conv, pad=2
  hipMemsetAsync(ws + OFF_SUM, 0, 1024, stream);
  conv(OFF_X4, OFF_PWD, OFF_XD, nullptr, true, 0, 162, 64, 32, 32, 32, 64, 64, 4, 4, 1, 2, 1);
  bnrelu(OFF_XD, gd, bd, 32, 4096);

  // cost head (Cout=1, scalar)
  conv5_kernel<<<(162 * 4096 + 255) / 256, 256, 0, stream>>>(
      (const unsigned short*)(ws + OFF_XD), w5, b5, (float*)(ws + OFF_COST));

  // DAP + softmax + embedding reduction -> final (B, 113, 64, 64) output
  dap_emb_kernel<<<(2 * 4096 + 255) / 256, 256, 0, stream>>>(
      (const float*)(ws + OFF_COST), (const unsigned short*)(ws + OFF_EMB),
      dapw, (float*)d_out);
}
